// VarianceAdaptor_31138512896007
// MI455X (gfx1250) — matrix operand order repaired and checked by reading, not compiled, not run on hardware
//
#include <hip/hip_runtime.h>
#include <hip/hip_bf16.h>

typedef __attribute__((ext_vector_type(16))) _Float16 v16h;
typedef __attribute__((ext_vector_type(8)))  _Float16 v8h;
typedef __attribute__((ext_vector_type(8)))  float    v8f;
typedef __attribute__((ext_vector_type(4)))  float    v4f;

#define BB     32
#define LL     512
#define TT     1600
#define HH     256
#define AST    264    // LDS activation slab stride (halves), padded vs 256
#define HST    260    // LDS epilogue stride (floats), padded vs 256
#define PW_ELEMS (3*8*16*32*16)   // 196608 f16 per packed conv weight

// ---------------------------------------------------------------------------
// Pack conv weight [3][H][H] f32 (k,ci,co) into WMMA B-fragment order, f16.
// pw[(((k*8+c)*16+nT)*512) + lane*16 + e] = w[k][c*32 + e + 16*(lane>>4)][nT*16 + (lane&15)]
// ---------------------------------------------------------------------------
__global__ __launch_bounds__(256) void pack_w_kernel(const float* __restrict__ w,
                                                     _Float16* __restrict__ pw) {
    int g    = blockIdx.x * 256 + threadIdx.x;         // 0..196607
    int e    = g & 15;
    int lane = (g >> 4) & 31;
    int nT   = (g >> 9) & 15;
    int c    = (g >> 13) & 7;
    int k    = g >> 16;
    int ci   = c * 32 + e + ((lane >> 4) << 4);
    int co   = nT * 16 + (lane & 15);
    pw[g] = (_Float16)w[((size_t)k * HH + ci) * HH + co];
}

// ---------------------------------------------------------------------------
// Fused conv1d(K=3,H->H) + bias + ReLU + LayerNorm (+ optional H->1 linear).
// Block: 256 threads (8 waves). Covers 16 time rows of one batch sample,
// all 256 output channels (each wave owns N-tiles {w, w+8}).
// ---------------------------------------------------------------------------
__global__ __launch_bounds__(256) void conv_ln_kernel(
        const float*    __restrict__ in,     // [B, Tlen, H]
        const _Float16* __restrict__ pw,     // packed f16 weights
        const float*    __restrict__ bias,   // [H]
        const float*    __restrict__ gamma,  // [H]
        const float*    __restrict__ beta,   // [H]
        float*          __restrict__ out_h,  // [B, Tlen, H] or nullptr
        const float*    __restrict__ wl,     // [H] or nullptr
        const float*    __restrict__ bl,     // [1] or nullptr
        float*          __restrict__ pred,   // [B, Tlen] or nullptr
        int Tlen)
{
    __shared__ _Float16 sA[18 * AST];
    __shared__ float    sH[16 * HST];
    __shared__ float    sSum[16][16];
    __shared__ float    sSsq[16][16];
    __shared__ float    sDot[16][16];
    __shared__ float    sMV[16][2];

    const int tid = threadIdx.x;
    const int b   = blockIdx.y;
    const int t0  = blockIdx.x * 16;

    // ---- Phase A: stage 18 halo rows x 256 ch, f32 -> f16 into LDS ----
    for (int i = tid; i < 18 * 64; i += 256) {
        int row  = i >> 6;
        int c4   = (i & 63) << 2;
        int srow = t0 - 1 + row;
        v4f v = {0.f, 0.f, 0.f, 0.f};
        if (srow >= 0 && srow < Tlen)
            v = *(const v4f*)(in + ((size_t)b * Tlen + srow) * HH + c4);
        _Float16* dst = &sA[row * AST + c4];
        dst[0] = (_Float16)v.x; dst[1] = (_Float16)v.y;
        dst[2] = (_Float16)v.z; dst[3] = (_Float16)v.w;
    }
    __syncthreads();

    const int wave  = tid >> 5;
    const int lane  = tid & 31;
    const int lrow  = lane & 15;
    const int lhalf = lane >> 4;
    const int n0 = wave, n1 = wave + 8;

    v8f acc0 = {}; v8f acc1 = {};

    // ---- Phase B: 3 taps x 8 K-chunks, 2 WMMAs each ----
    for (int k = 0; k < 3; ++k) {
        #pragma unroll
        for (int c = 0; c < 8; ++c) {
            // A fragment (16x32 f16) from LDS in contiguous k order (k = 16*lhalf + e), same as packed B
            int abase = (lrow + k) * AST + c * 32 + lhalf * 16;
            v8h alo = *(const v8h*)(&sA[abase]);
            v8h ahi = *(const v8h*)(&sA[abase + 8]);
            v16h a;
            #pragma unroll
            for (int e2 = 0; e2 < 8; ++e2) { a[e2] = alo[e2]; a[e2 + 8] = ahi[e2]; }
            // B fragments: contiguous 32B per lane in packed layout
            const _Float16* pb = pw + (size_t)((k * 8 + c) * 16) * 512 + lane * 16;
            v16h b0 = *(const v16h*)(pb + (size_t)n0 * 512);
            v16h b1 = *(const v16h*)(pb + (size_t)n1 * 512);
            acc0 = __builtin_amdgcn_wmma_f32_16x16x32_f16(false, a, false, b0,
                                                          (short)0, acc0, false, false);
            acc1 = __builtin_amdgcn_wmma_f32_16x16x32_f16(false, a, false, b1,
                                                          (short)0, acc1, false, false);
        }
    }

    // ---- Phase C: bias + ReLU into LDS (C/D layout: N=lane&15, M=r+8*(lane>>4)) ----
    #pragma unroll
    for (int r = 0; r < 8; ++r) {
        int m = r + lhalf * 8;
        int ch0 = n0 * 16 + lrow;
        float v0 = acc0[r] + bias[ch0];
        sH[m * HST + ch0] = v0 > 0.f ? v0 : 0.f;
        int ch1 = n1 * 16 + lrow;
        float v1 = acc1[r] + bias[ch1];
        sH[m * HST + ch1] = v1 > 0.f ? v1 : 0.f;
    }
    __syncthreads();

    // ---- Phase D: LayerNorm over H per row (16 threads per row) ----
    const int row = tid >> 4, sub = tid & 15;
    float s = 0.f, sq = 0.f;
    #pragma unroll
    for (int j = 0; j < 16; ++j) {
        float v = sH[row * HST + sub * 16 + j];
        s += v; sq += v * v;
    }
    sSum[row][sub] = s; sSsq[row][sub] = sq;
    __syncthreads();
    if (sub == 0) {
        float ts = 0.f, tq = 0.f;
        #pragma unroll
        for (int j = 0; j < 16; ++j) { ts += sSum[row][j]; tq += sSsq[row][j]; }
        float mean = ts * (1.f / HH);
        float var  = tq * (1.f / HH) - mean * mean;
        sMV[row][0] = mean;
        sMV[row][1] = rsqrtf(var + 1e-5f);
    }
    __syncthreads();

    const float mean = sMV[row][0], rstd = sMV[row][1];
    const int t = t0 + row;
    float dot = 0.f;
    #pragma unroll
    for (int j = 0; j < 16; ++j) {
        int ch = sub * 16 + j;
        float v = (sH[row * HST + ch] - mean) * rstd * gamma[ch] + beta[ch];
        if (out_h) out_h[((size_t)b * Tlen + t) * HH + ch] = v;
        if (wl) dot += v * wl[ch];
    }
    if (wl) {  // uniform branch
        sDot[row][sub] = dot;
        __syncthreads();
        if (sub == 0) {
            float td = 0.f;
            #pragma unroll
            for (int j = 0; j < 16; ++j) td += sDot[row][j];
            pred[(size_t)b * Tlen + t] = td + bl[0];
        }
    }
}

// ---------------------------------------------------------------------------
// Inclusive scan of durations per batch row (L=512).
// ---------------------------------------------------------------------------
__global__ __launch_bounds__(512) void cumsum_kernel(const int* __restrict__ dur,
                                                     int* __restrict__ cums) {
    __shared__ int s[LL];
    const int b = blockIdx.x, tid = threadIdx.x;
    s[tid] = dur[b * LL + tid];
    __syncthreads();
    for (int off = 1; off < LL; off <<= 1) {
        int v = (tid >= off) ? s[tid - off] : 0;
        __syncthreads();
        s[tid] += v;
        __syncthreads();
    }
    cums[b * LL + tid] = s[tid];
}

// ---------------------------------------------------------------------------
// Length regulate: frame t copies phoneme idx = searchsorted(cums, t, 'right').
// ---------------------------------------------------------------------------
__global__ __launch_bounds__(256) void regulate_kernel(const float* __restrict__ x,
                                                       const int*   __restrict__ cums,
                                                       float* __restrict__ xmel,
                                                       float* __restrict__ melpos) {
    const int b = blockIdx.y, t = blockIdx.x, ch = threadIdx.x;
    const int* c = cums + b * LL;
    const bool mask = t < c[LL - 1];
    int lo = 0, hi = LL;
    while (lo < hi) { int mid = (lo + hi) >> 1; if (c[mid] <= t) lo = mid + 1; else hi = mid; }
    int idx = lo < (LL - 1) ? lo : (LL - 1);
    float v = mask ? x[((size_t)b * LL + idx) * HH + ch] : 0.f;
    xmel[((size_t)b * TT + t) * HH + ch] = v;
    if (ch == 0) melpos[(size_t)b * TT + t] = mask ? (float)(t + 1) : 0.f;
}

// ---------------------------------------------------------------------------
// Bucketize target (searchsorted side='left' over 255 bins) and add embedding row.
// ---------------------------------------------------------------------------
__global__ __launch_bounds__(256) void add_emb_kernel(const float* __restrict__ bins,
                                                      const float* __restrict__ tgt,
                                                      const float* __restrict__ emb,
                                                      float* __restrict__ xmel) {
    const int b = blockIdx.y, t = blockIdx.x, ch = threadIdx.x;
    const float v = tgt[(size_t)b * TT + t];
    int lo = 0, hi = 255;
    while (lo < hi) { int mid = (lo + hi) >> 1; if (bins[mid] < v) lo = mid + 1; else hi = mid; }
    xmel[((size_t)b * TT + t) * HH + ch] += emb[(size_t)lo * HH + ch];
}

// ---------------------------------------------------------------------------
extern "C" void kernel_launch(void* const* d_in, const int* in_sizes, int n_in,
                              void* d_out, int out_size, void* d_ws, size_t ws_size,
                              hipStream_t stream) {
    (void)in_sizes; (void)n_in; (void)out_size; (void)ws_size;

    const float* x       = (const float*)d_in[0];
    const int*   dur_tgt = (const int*)d_in[1];
    const float* pitchT  = (const float*)d_in[2];
    const float* energyT = (const float*)d_in[3];
    // d_in[4] = max_length (T, known constant)
    // predictor params: dur 5..14, pitch 15..24, energy 25..34
    const float* pitch_bins  = (const float*)d_in[35];
    const float* energy_bins = (const float*)d_in[36];
    const float* pitch_emb   = (const float*)d_in[37];
    const float* energy_emb  = (const float*)d_in[38];

    // workspace layout
    char*     ws  = (char*)d_ws;
    _Float16* pw  = (_Float16*)ws;                                  // 6 * 196608 f16
    int*      cums = (int*)(ws + (size_t)6 * PW_ELEMS * 2);         // B*L i32
    float*    h1   = (float*)(ws + (size_t)6 * PW_ELEMS * 2 + (size_t)BB * LL * 4);

    // output layout (all float)
    float* out         = (float*)d_out;
    float* xmel        = out;                                       // [B,T,H]
    float* dur_pred    = xmel + (size_t)BB * TT * HH;               // [B,L]
    float* pitch_pred  = dur_pred + (size_t)BB * LL;                // [B,T]
    float* energy_pred = pitch_pred + (size_t)BB * TT;              // [B,T]
    float* mel_pos     = energy_pred + (size_t)BB * TT;             // [B,T]

    // 1) pack the six conv weights into WMMA-fragment order
    const int widx[6] = {5, 9, 15, 19, 25, 29};
    for (int i = 0; i < 6; ++i)
        pack_w_kernel<<<PW_ELEMS / 256, 256, 0, stream>>>(
            (const float*)d_in[widx[i]], pw + (size_t)i * PW_ELEMS);

    const dim3 gDur(LL / 16, BB);
    const dim3 gMel(TT / 16, BB);
    const dim3 gRow(TT, BB);

    // 2) duration predictor on x  (Tlen = L)
    conv_ln_kernel<<<gDur, 256, 0, stream>>>(x, pw,
        (const float*)d_in[6], (const float*)d_in[7], (const float*)d_in[8],
        h1, nullptr, nullptr, nullptr, LL);
    conv_ln_kernel<<<gDur, 256, 0, stream>>>(h1, pw + (size_t)1 * PW_ELEMS,
        (const float*)d_in[10], (const float*)d_in[11], (const float*)d_in[12],
        nullptr, (const float*)d_in[13], (const float*)d_in[14], dur_pred, LL);

    // 3) length regulate -> x_mel (in d_out), mel_pos
    cumsum_kernel<<<BB, LL, 0, stream>>>(dur_tgt, cums);
    regulate_kernel<<<gRow, 256, 0, stream>>>(x, cums, xmel, mel_pos);

    // 4) pitch predictor on x_mel (pre-embedding)
    conv_ln_kernel<<<gMel, 256, 0, stream>>>(xmel, pw + (size_t)2 * PW_ELEMS,
        (const float*)d_in[16], (const float*)d_in[17], (const float*)d_in[18],
        h1, nullptr, nullptr, nullptr, TT);
    conv_ln_kernel<<<gMel, 256, 0, stream>>>(h1, pw + (size_t)3 * PW_ELEMS,
        (const float*)d_in[20], (const float*)d_in[21], (const float*)d_in[22],
        nullptr, (const float*)d_in[23], (const float*)d_in[24], pitch_pred, TT);

    // 5) x_mel += pitch_emb[bucketize(pitch_target)]
    add_emb_kernel<<<gRow, 256, 0, stream>>>(pitch_bins, pitchT, pitch_emb, xmel);

    // 6) energy predictor on updated x_mel
    conv_ln_kernel<<<gMel, 256, 0, stream>>>(xmel, pw + (size_t)4 * PW_ELEMS,
        (const float*)d_in[26], (const float*)d_in[27], (const float*)d_in[28],
        h1, nullptr, nullptr, nullptr, TT);
    conv_ln_kernel<<<gMel, 256, 0, stream>>>(h1, pw + (size_t)5 * PW_ELEMS,
        (const float*)d_in[30], (const float*)d_in[31], (const float*)d_in[32],
        nullptr, (const float*)d_in[33], (const float*)d_in[34], energy_pred, TT);

    // 7) x_mel += energy_emb[bucketize(energy_target)]
    add_emb_kernel<<<gRow, 256, 0, stream>>>(energy_bins, energyT, energy_emb, xmel);
}
